// QThreeLayer_47725676593744
// MI455X (gfx1250) — 8-bit matrix operand order repaired and checked by reading, compiled, not run on hardware
//
#include <hip/hip_runtime.h>
#include <math.h>


typedef __attribute__((ext_vector_type(8))) int v8i;
typedef int v4i_vec __attribute__((vector_size(16)));

#define QEPS 1e-8f
#define QN   31
#define QLO  (-32)
#define QHI  31

#if __has_builtin(__builtin_amdgcn_global_load_async_to_lds_b128) && \
    __has_builtin(__builtin_amdgcn_s_wait_asynccnt)
#define USE_ASYNC_COPY 1
#else
#define USE_ASYNC_COPY 0
#endif

#if USE_ASYNC_COPY
#define GLOBAL_AS __attribute__((address_space(1)))
#define LDS_AS    __attribute__((address_space(3)))
__device__ __forceinline__ void async_b128(const void* g, void* l) {
    __builtin_amdgcn_global_load_async_to_lds_b128(
        (GLOBAL_AS v4i_vec*)g, (LDS_AS v4i_vec*)l, 0, 0);
}
#endif

__device__ __forceinline__ int clampi(int v, int lo, int hi) {
    return v < lo ? lo : (v > hi ? hi : v);
}

__global__ void init_amax_kernel(unsigned int* amax) {
    if (threadIdx.x < 4) amax[threadIdx.x] = 0u;
}

__global__ __launch_bounds__(256) void absmax_f32_kernel(
    const float* __restrict__ x, long long n, unsigned int* __restrict__ amax)
{
    __shared__ float red[256];
    float m = 0.0f;
    for (long long i = (long long)blockIdx.x * blockDim.x + threadIdx.x; i < n;
         i += (long long)gridDim.x * blockDim.x)
        m = fmaxf(m, fabsf(x[i]));
    red[threadIdx.x] = m;
    __syncthreads();
    for (int s = 128; s > 0; s >>= 1) {
        if ((int)threadIdx.x < s)
            red[threadIdx.x] = fmaxf(red[threadIdx.x], red[threadIdx.x + s]);
        __syncthreads();
    }
    if (threadIdx.x == 0) atomicMax(amax, __float_as_uint(red[0]));
}

__global__ __launch_bounds__(256) void quant_act_kernel(
    const float* __restrict__ a, signed char* __restrict__ q, long long n4,
    const unsigned int* __restrict__ amax)
{
    const float s   = fmaxf(__uint_as_float(*amax) * (1.0f / QN), QEPS);
    const float inv = 1.0f / s;
    const float4* a4 = (const float4*)a;
    char4*       q4 = (char4*)q;
    for (long long i = (long long)blockIdx.x * blockDim.x + threadIdx.x; i < n4;
         i += (long long)gridDim.x * blockDim.x) {
        float4 v = a4[i];
        char4  c;
        c.x = (signed char)clampi(__float2int_rn(v.x * inv), QLO, QHI);
        c.y = (signed char)clampi(__float2int_rn(v.y * inv), QLO, QHI);
        c.z = (signed char)clampi(__float2int_rn(v.z * inv), QLO, QHI);
        c.w = (signed char)clampi(__float2int_rn(v.w * inv), QLO, QHI);
        q4[i] = c;
    }
}

__global__ __launch_bounds__(256) void quant_weights_kernel(
    const float* __restrict__ W, signed char* __restrict__ Wq,
    float* __restrict__ wsRow, int K)
{
    const int o = blockIdx.x;
    const float* w = W + (long long)o * K;
    __shared__ float red[256];
    __shared__ float sws;
    float m = 0.0f;
    for (int k = threadIdx.x; k < K; k += blockDim.x) m = fmaxf(m, fabsf(w[k]));
    red[threadIdx.x] = m;
    __syncthreads();
    for (int s = 128; s > 0; s >>= 1) {
        if ((int)threadIdx.x < s)
            red[threadIdx.x] = fmaxf(red[threadIdx.x], red[threadIdx.x + s]);
        __syncthreads();
    }
    if (threadIdx.x == 0) {
        sws = fmaxf(red[0] * (1.0f / QN), QEPS);
        wsRow[o] = sws;
    }
    __syncthreads();
    const float inv = 1.0f / sws;
    signed char* wq = Wq + (long long)o * K;
    for (int k = threadIdx.x; k < K; k += blockDim.x)
        wq[k] = (signed char)clampi(__float2int_rn(w[k] * inv), QLO, QHI);
}

#define BM 128
#define BN 128
#define BK 64

__global__ __launch_bounds__(256) void gemm_q8_wmma_kernel(
    const signed char* __restrict__ Aq,
    const signed char* __restrict__ Wq,
    const float* __restrict__ wsRow,
    const float* __restrict__ bias,
    float* __restrict__ H,
    const unsigned int* __restrict__ actAmax,
    unsigned int* __restrict__ nextAmax,
    int M, int N, int K)
{
    __shared__ __align__(16) signed char As[2][BM][BK];
    __shared__ __align__(16) signed char Bs[2][BN][BK];
    __shared__ float red[256];

    const int tid  = threadIdx.x;
    const int wave = tid >> 5;
    const int lane = tid & 31;
    const int wm   = wave >> 1;
    const int wn   = wave & 1;
    const int l    = lane & 15;
    const int hi   = lane >> 4;
    const int m0   = blockIdx.y * BM;
    const int n0   = blockIdx.x * BN;

    const int r0   = tid >> 2;
    const int r1   = r0 + 64;
    const int cOfs = (tid & 3) * 16;

    const signed char* aRow0 = Aq + (long long)(m0 + r0) * K + cOfs;
    const signed char* aRow1 = Aq + (long long)(m0 + r1) * K + cOfs;
    const signed char* bRow0 = Wq + (long long)(n0 + r0) * K + cOfs;
    const signed char* bRow1 = Wq + (long long)(n0 + r1) * K + cOfs;

    v8i acc[2][4];
#pragma unroll
    for (int i = 0; i < 2; ++i)
#pragma unroll
        for (int j = 0; j < 4; ++j) acc[i][j] = v8i{};

    const int nIter = K / BK;

#if USE_ASYNC_COPY
    async_b128(aRow0, &As[0][r0][cOfs]);
    async_b128(aRow1, &As[0][r1][cOfs]);
    async_b128(bRow0, &Bs[0][r0][cOfs]);
    async_b128(bRow1, &Bs[0][r1][cOfs]);
#else
    int4 pa0 = *(const int4*)aRow0;
    int4 pa1 = *(const int4*)aRow1;
    int4 pb0 = *(const int4*)bRow0;
    int4 pb1 = *(const int4*)bRow1;
#endif

    for (int it = 0; it < nIter; ++it) {
        const int cur = it & 1;
        const int nk  = (it + 1) * BK;

#if USE_ASYNC_COPY
        __builtin_amdgcn_s_wait_asynccnt(0);
        __syncthreads();
        if (it + 1 < nIter) {
            async_b128(aRow0 + nk, &As[cur ^ 1][r0][cOfs]);
            async_b128(aRow1 + nk, &As[cur ^ 1][r1][cOfs]);
            async_b128(bRow0 + nk, &Bs[cur ^ 1][r0][cOfs]);
            async_b128(bRow1 + nk, &Bs[cur ^ 1][r1][cOfs]);
        }
#else
        *(int4*)&As[cur][r0][cOfs] = pa0;
        *(int4*)&As[cur][r1][cOfs] = pa1;
        *(int4*)&Bs[cur][r0][cOfs] = pb0;
        *(int4*)&Bs[cur][r1][cOfs] = pb1;
        __syncthreads();
        if (it + 1 < nIter) {
            pa0 = *(const int4*)(aRow0 + nk);
            pa1 = *(const int4*)(aRow1 + nk);
            pb0 = *(const int4*)(bRow0 + nk);
            pb1 = *(const int4*)(bRow1 + nk);
        }
#endif

        v8i afrag[2];
#pragma unroll
        for (int mt = 0; mt < 2; ++mt) {
            const int m = wm * 32 + mt * 16 + l;
#pragma unroll
            for (int g = 0; g < 2; ++g) {
                int4 d = *(const int4*)&As[cur][m][g * 32 + hi * 16];
                afrag[mt][4 * g]     = d.x;
                afrag[mt][4 * g + 1] = d.y;
                afrag[mt][4 * g + 2] = d.z;
                afrag[mt][4 * g + 3] = d.w;
            }
        }
        v8i bfrag[4];
#pragma unroll
        for (int nt = 0; nt < 4; ++nt) {
            const int n = wn * 64 + nt * 16 + l;
            int4 d0 = *(const int4*)&Bs[cur][n][hi * 16];
            int4 d1 = *(const int4*)&Bs[cur][n][32 + hi * 16];
            bfrag[nt][0] = d0.x; bfrag[nt][1] = d0.y;
            bfrag[nt][2] = d0.z; bfrag[nt][3] = d0.w;
            bfrag[nt][4] = d1.x; bfrag[nt][5] = d1.y;
            bfrag[nt][6] = d1.z; bfrag[nt][7] = d1.w;
        }
#pragma unroll
        for (int mt = 0; mt < 2; ++mt)
#pragma unroll
            for (int nt = 0; nt < 4; ++nt)
                acc[mt][nt] = __builtin_amdgcn_wmma_i32_16x16x64_iu8(
                     true, afrag[mt],  true, bfrag[nt],
                    acc[mt][nt],  false,  false);
    }

    const float s = fmaxf(__uint_as_float(*actAmax) * (1.0f / QN), QEPS);
    float tmax = 0.0f;
#pragma unroll
    for (int mt = 0; mt < 2; ++mt) {
        const int mbase = m0 + wm * 32 + mt * 16 + hi * 8;
#pragma unroll
        for (int nt = 0; nt < 4; ++nt) {
            const int   ncol = n0 + wn * 64 + nt * 16 + l;
            const float bs   = wsRow[ncol] * s;
            const float bi   =
                (float)clampi(__float2int_rn(bias[ncol] / bs), QLO, QHI);
#pragma unroll
            for (int j = 0; j < 8; ++j) {
                float h = ((float)acc[mt][nt][j] + bi) * bs;
                h = fmaxf(h, 0.0f);
                H[(long long)(mbase + j) * N + ncol] = h;
                tmax = fmaxf(tmax, h);
            }
        }
    }
    red[tid] = tmax;
    __syncthreads();
    for (int sft = 128; sft > 0; sft >>= 1) {
        if (tid < sft) red[tid] = fmaxf(red[tid], red[tid + sft]);
        __syncthreads();
    }
    if (tid == 0) atomicMax(nextAmax, __float_as_uint(red[0]));
}

__global__ __launch_bounds__(256) void final_softmax_kernel(
    const signed char* __restrict__ Aq,
    const signed char* __restrict__ Wq4,
    const float* __restrict__ ws4,
    const float* __restrict__ b4,
    const unsigned int* __restrict__ actAmax,
    float* __restrict__ out,
    int M, int K)
{
    __shared__ __align__(16) signed char Wl[5 * 2048];
    const int nInt = 5 * 2048 / 4;
    for (int i = threadIdx.x; i < nInt; i += blockDim.x)
        ((int*)Wl)[i] = ((const int*)Wq4)[i];
    __syncthreads();

    const float s = fmaxf(__uint_as_float(*actAmax) * (1.0f / QN), QEPS);
    const int row = blockIdx.x * blockDim.x + threadIdx.x;
    if (row >= M) return;

    const signed char* a = Aq + (long long)row * K;
    int acc[5] = {0, 0, 0, 0, 0};
    for (int k = 0; k < K; k += 4) {
        char4 av = *(const char4*)&a[k];
#pragma unroll
        for (int c = 0; c < 5; ++c) {
            char4 wv = *(const char4*)&Wl[c * K + k];
            acc[c] += (int)av.x * wv.x + (int)av.y * wv.y +
                      (int)av.z * wv.z + (int)av.w * wv.w;
        }
    }
    float logit[5];
    float mx = -1e30f;
#pragma unroll
    for (int c = 0; c < 5; ++c) {
        const float bs = ws4[c] * s;
        const float bi = (float)clampi(__float2int_rn(b4[c] / bs), QLO, QHI);
        logit[c] = ((float)acc[c] + bi) * bs;
        mx = fmaxf(mx, logit[c]);
    }
    float sum = 0.0f;
#pragma unroll
    for (int c = 0; c < 5; ++c) { logit[c] = expf(logit[c] - mx); sum += logit[c]; }
    const float inv = 1.0f / sum;
#pragma unroll
    for (int c = 0; c < 5; ++c) out[(long long)row * 5 + c] = logit[c] * inv;
}

extern "C" void kernel_launch(void* const* d_in, const int* in_sizes, int n_in,
                              void* d_out, int out_size, void* d_ws, size_t ws_size,
                              hipStream_t stream) {
    const int B = 16384, D0 = 2048, D1 = 4096, D2 = 4096, D3 = 2048, C = 5;

    const float* x  = (const float*)d_in[0];
    const float* W1 = (const float*)d_in[1];
    const float* b1 = (const float*)d_in[2];
    const float* W2 = (const float*)d_in[3];
    const float* b2 = (const float*)d_in[4];
    const float* W3 = (const float*)d_in[5];
    const float* b3 = (const float*)d_in[6];
    const float* W4 = (const float*)d_in[7];
    const float* b4 = (const float*)d_in[8];

    char* ws = (char*)d_ws;
    unsigned int* amax = (unsigned int*)ws;
    size_t off = 256;
    float* actF = (float*)(ws + off);        off += (size_t)B * 4096 * sizeof(float);
    signed char* actQ = (signed char*)(ws + off); off += (size_t)B * 4096;
    float* wsRow = (float*)(ws + off);       off += (size_t)4096 * sizeof(float);
    signed char* wq = (signed char*)(ws + off);   off += (size_t)4096 * 4096;
    (void)ws_size; (void)n_in; (void)in_sizes; (void)out_size;

    float* out = (float*)d_out;

    init_amax_kernel<<<1, 32, 0, stream>>>(amax);

    absmax_f32_kernel<<<1024, 256, 0, stream>>>(x, (long long)B * D0, amax + 0);
    quant_act_kernel<<<2048, 256, 0, stream>>>(x, actQ, (long long)B * D0 / 4, amax + 0);
    quant_weights_kernel<<<D1, 256, 0, stream>>>(W1, wq, wsRow, D0);
    gemm_q8_wmma_kernel<<<dim3(D1 / BN, B / BM), 256, 0, stream>>>(
        actQ, wq, wsRow, b1, actF, amax + 0, amax + 1, B, D1, D0);

    quant_act_kernel<<<2048, 256, 0, stream>>>(actF, actQ, (long long)B * D1 / 4, amax + 1);
    quant_weights_kernel<<<D2, 256, 0, stream>>>(W2, wq, wsRow, D1);
    gemm_q8_wmma_kernel<<<dim3(D2 / BN, B / BM), 256, 0, stream>>>(
        actQ, wq, wsRow, b2, actF, amax + 1, amax + 2, B, D2, D1);

    quant_act_kernel<<<2048, 256, 0, stream>>>(actF, actQ, (long long)B * D2 / 4, amax + 2);
    quant_weights_kernel<<<D3, 256, 0, stream>>>(W3, wq, wsRow, D2);
    gemm_q8_wmma_kernel<<<dim3(D3 / BN, B / BM), 256, 0, stream>>>(
        actQ, wq, wsRow, b3, actF, amax + 2, amax + 3, B, D3, D2);

    quant_act_kernel<<<2048, 256, 0, stream>>>(actF, actQ, (long long)B * D3 / 4, amax + 3);
    quant_weights_kernel<<<C, 256, 0, stream>>>(W4, wq, wsRow, D3);
    final_softmax_kernel<<<(B + 255) / 256, 256, 0, stream>>>(
        actQ, wq, wsRow, b4, amax + 3, out, B, D3);
}
